// Net_15479062135163
// MI455X (gfx1250) — matrix operand order repaired and checked by reading, not compiled, not run on hardware
//
#include <hip/hip_runtime.h>

#define NNODE 8192
#define FIN   128
#define FHID  64
#define FOUT  2

typedef __attribute__((ext_vector_type(16))) __bf16 v16bf;
typedef __attribute__((ext_vector_type(8)))  __bf16 v8bf;
typedef __attribute__((ext_vector_type(8)))  float  v8f;

// ---------------------------------------------------------------------------
// A-fragment: 16x32 bf16 tile of adj (row-major fp32 in memory), converted to
// bf16 in registers. Contiguous-K layout used here (same K order as B):
//   lanes 0-15 : row = lane,    elems 0..15 -> K = kb+0..15
//   lanes 16-31: row = lane-16, elems 0..15 -> K = kb+16..31
// ---------------------------------------------------------------------------
__device__ __forceinline__ v16bf load_a_frag_f32(const float* __restrict__ arow,
                                                 int kb, int khalf) {
  const float* p = arow + kb + khalf * 16;
  float4 f0 = *(const float4*)(p + 0);
  float4 f1 = *(const float4*)(p + 4);
  float4 f2 = *(const float4*)(p + 8);
  float4 f3 = *(const float4*)(p + 12);
  v16bf a;
  a[0]  = (__bf16)f0.x; a[1]  = (__bf16)f0.y; a[2]  = (__bf16)f0.z; a[3]  = (__bf16)f0.w;
  a[4]  = (__bf16)f1.x; a[5]  = (__bf16)f1.y; a[6]  = (__bf16)f1.z; a[7]  = (__bf16)f1.w;
  a[8]  = (__bf16)f2.x; a[9]  = (__bf16)f2.y; a[10] = (__bf16)f2.z; a[11] = (__bf16)f2.w;
  a[12] = (__bf16)f3.x; a[13] = (__bf16)f3.y; a[14] = (__bf16)f3.z; a[15] = (__bf16)f3.w;
  return a;
}

// ---------------------------------------------------------------------------
// B-fragment: 32x16 bf16. Y is stored transposed (Yt[col][k], row-major by col)
// so each lane's 16 K-values are contiguous:
//   lanes 0-15 : col = lane,    K = kb+0..15
//   lanes 16-31: col = lane-16, K = kb+16..31
// ---------------------------------------------------------------------------
__device__ __forceinline__ v16bf load_b_frag(const __bf16* __restrict__ ycol,
                                             int kb, int khalf) {
  const __bf16* p = ycol + kb + khalf * 16;
  v8bf b0 = *(const v8bf*)(p);
  v8bf b1 = *(const v8bf*)(p + 8);
  v16bf b;
#pragma unroll
  for (int i = 0; i < 8; ++i) { b[i] = b0[i]; b[8 + i] = b1[i]; }
  return b;
}

// ---------------------------------------------------------------------------
// Kernel 1: row sums of adj -> dinv[row] = rsqrt(sum + eps)
// ---------------------------------------------------------------------------
__global__ void __launch_bounds__(256) gcn_rowsum(const float* __restrict__ adj,
                                                  float* __restrict__ dinv) {
  const int row = blockIdx.x;
  const float4* a = (const float4*)(adj + (size_t)row * NNODE);
  float s = 0.f;
  for (int i = threadIdx.x; i < NNODE / 4; i += 256) {
    float4 v = a[i];
    s += (v.x + v.y) + (v.z + v.w);
  }
#pragma unroll
  for (int off = 16; off > 0; off >>= 1) s += __shfl_down(s, off, 32);
  __shared__ float red[8];
  if ((threadIdx.x & 31) == 0) red[threadIdx.x >> 5] = s;
  __syncthreads();
  if (threadIdx.x == 0) {
    float t = 0.f;
#pragma unroll
    for (int w = 0; w < 8; ++w) t += red[w];
    dinv[row] = rsqrtf(t + 1e-6f);
  }
}

// ---------------------------------------------------------------------------
// Kernel 2: Yt1[c][j] = bf16( d_j * (x @ W1)[j][c] )   (transposed, bf16)
// ---------------------------------------------------------------------------
__global__ void __launch_bounds__(64) gcn_xw1(const float* __restrict__ x,
                                              const float* __restrict__ W1,
                                              const float* __restrict__ dinv,
                                              __bf16* __restrict__ yt1) {
  const int j = blockIdx.x;
  const int c = threadIdx.x;            // 0..63
  __shared__ float xs[FIN];
#pragma unroll
  for (int i = c; i < FIN; i += 64) xs[i] = x[(size_t)j * FIN + i];
  __syncthreads();
  float acc = 0.f;
#pragma unroll 8
  for (int k = 0; k < FIN; ++k) acc = fmaf(xs[k], W1[k * FHID + c], acc);
  yt1[(size_t)c * NNODE + j] = (__bf16)(acc * dinv[j]);
}

// ---------------------------------------------------------------------------
// Kernel 3 (the hot one): C = A @ Yt^T, then out = post(d_i * C + b).
// Waves are arranged as NC column-tiles x NK K-splits per 16-row block:
//   wave w: ct = w % NC (16-col tile), ks = w / NC (K range, disjoint).
// Each wave runs the bf16-WMMA loop over its K slice with dual accumulators,
// dumps its partial 16x16 tile to LDS, then the block reduces over ks and
// applies dinv/bias/relu. K-splitting multiplies in-flight loads (latency
// hiding for the 256 MB/layer stream of adj) without changing HBM traffic.
// ---------------------------------------------------------------------------
template <int NC, int NK>
__global__ void __launch_bounds__(32 * NC * NK) gcn_agg_wmma(
    const float* __restrict__ adjf, const __bf16* __restrict__ yt,
    const float* __restrict__ dinv, const float* __restrict__ bias,
    float* __restrict__ out, int do_relu, int out_cols, int out_ld) {
  const int lane  = threadIdx.x & 31;
  const int wave  = threadIdx.x >> 5;
  const int ct    = wave % NC;        // column tile
  const int ks    = wave / NC;        // K split
  const int r0    = blockIdx.x * 16;
  const int khalf = lane >> 4;
  const int l15   = lane & 15;

  const float*  arow = adjf + (size_t)(r0 + l15) * NNODE;
  const __bf16* ycol = yt + (size_t)(ct * 16 + l15) * NNODE;

  constexpr int KSLICE = NNODE / NK;
  const int k0 = ks * KSLICE;

  v8f acc0 = {};
  v8f acc1 = {};
  for (int kb = k0; kb < k0 + KSLICE; kb += 64) {
    v16bf a0 = load_a_frag_f32(arow, kb, khalf);
    v16bf b0 = load_b_frag(ycol, kb, khalf);
    acc0 = __builtin_amdgcn_wmma_f32_16x16x32_bf16(false, a0, false, b0,
                                                   (short)0, acc0, false, false);
    v16bf a1 = load_a_frag_f32(arow, kb + 32, khalf);
    v16bf b1 = load_b_frag(ycol, kb + 32, khalf);
    acc1 = __builtin_amdgcn_wmma_f32_16x16x32_bf16(false, a1, false, b1,
                                                   (short)0, acc1, false, false);
  }
  v8f acc = acc0 + acc1;

  // Partial-tile reduction over K splits via LDS.
  // C/D layout: lane holds N = lane&15; VGPR v holds M = (lane>>4)*8 + v.
  __shared__ float part[NK * NC * 256];
  {
    float* p = &part[((ks * NC + ct) * 256) + lane * 8];
    *(float4*)(p + 0) = make_float4(acc[0], acc[1], acc[2], acc[3]);
    *(float4*)(p + 4) = make_float4(acc[4], acc[5], acc[6], acc[7]);
  }
  __syncthreads();

  for (int e = threadIdx.x; e < NC * 256; e += 32 * NC * NK) {
    const int ect  = e >> 8;          // column tile
    const int idx  = e & 255;         // lane*8 + v
    const int el   = idx >> 3;
    const int ev   = idx & 7;
    float s = 0.f;
#pragma unroll
    for (int k = 0; k < NK; ++k) s += part[(k * NC + ect) * 256 + idx];
    const int row = r0 + (el >> 4) * 8 + ev;
    const int col = ect * 16 + (el & 15);
    if (col < out_cols) {
      float val = s * dinv[row] + bias[col];
      if (do_relu) val = fmaxf(val, 0.0f);
      out[(size_t)row * out_ld + col] = val;
    }
  }
}

// ---------------------------------------------------------------------------
// Kernel 4: Yt2[c][j] = bf16( d_j * (h @ W2)[j][c] ), cols 2..15 zero-padded
// ---------------------------------------------------------------------------
__global__ void __launch_bounds__(256) gcn_hw2(const float* __restrict__ h,
                                               const float* __restrict__ W2,
                                               const float* __restrict__ dinv,
                                               __bf16* __restrict__ yt2) {
  const int j = blockIdx.x * 256 + threadIdx.x;
  if (j >= NNODE) return;
  float a0 = 0.f, a1 = 0.f;
#pragma unroll 8
  for (int k = 0; k < FHID; ++k) {
    float hv = h[(size_t)j * FHID + k];
    a0 = fmaf(hv, W2[k * FOUT + 0], a0);
    a1 = fmaf(hv, W2[k * FOUT + 1], a1);
  }
  const float d = dinv[j];
  yt2[(size_t)0 * NNODE + j] = (__bf16)(a0 * d);
  yt2[(size_t)1 * NNODE + j] = (__bf16)(a1 * d);
#pragma unroll
  for (int c = 2; c < 16; ++c) yt2[(size_t)c * NNODE + j] = (__bf16)0.f;
}

// ---------------------------------------------------------------------------
extern "C" void kernel_launch(void* const* d_in, const int* in_sizes, int n_in,
                              void* d_out, int out_size, void* d_ws, size_t ws_size,
                              hipStream_t stream) {
  const float* x   = (const float*)d_in[0];   // [8192,128]
  const float* adj = (const float*)d_in[1];   // [8192,8192]
  const float* W1  = (const float*)d_in[2];   // [128,64]
  const float* b1  = (const float*)d_in[3];   // [64]
  const float* W2  = (const float*)d_in[4];   // [64,2]
  const float* b2  = (const float*)d_in[5];   // [2]
  float* out = (float*)d_out;                 // [8192,2]

  char* ws = (char*)d_ws;
  float*  dinv = (float*)(ws);                                          // 32 KB
  __bf16* yt1  = (__bf16*)(ws + (size_t)(32 << 10));                    // 1 MB  [64,8192]
  float*  h    = (float*)(ws + (size_t)(32 << 10) + (1u << 20));        // 2 MB  [8192,64]
  __bf16* yt2  = (__bf16*)(ws + (size_t)(32 << 10) + (3u << 20));       // 256KB [16,8192]

  // 1) degrees -> d^-1/2
  gcn_rowsum<<<NNODE, 256, 0, stream>>>(adj, dinv);
  // 2) Yt1 = (d_j * (x @ W1))^T in bf16
  gcn_xw1<<<NNODE, 64, 0, stream>>>(x, W1, dinv, yt1);
  // 3) h = relu(d_i * (A @ Y1) + b1) -- WMMA bf16, 4 col-tiles x 4 K-splits
  gcn_agg_wmma<4, 4><<<NNODE / 16, 32 * 4 * 4, 0, stream>>>(
      adj, yt1, dinv, b1, h, /*relu=*/1, FHID, FHID);
  // 4) Yt2 = (d_j * (h @ W2))^T padded to 16 cols, bf16
  gcn_hw2<<<NNODE / 256, 256, 0, stream>>>(h, W2, dinv, yt2);
  // 5) out = d_i * (A @ Y2) + b2     -- WMMA bf16, 1 col-tile x 8 K-splits
  gcn_agg_wmma<1, 8><<<NNODE / 16, 32 * 1 * 8, 0, stream>>>(
      adj, yt2, dinv, b2, out, /*relu=*/0, FOUT, FOUT);
}
